// StrideMHAModule_4466765988501
// MI455X (gfx1250) — compile-verified, matrix operand order checked by reading, not run on hardware
//
#include <hip/hip_runtime.h>
#include <hip/hip_fp16.h>

typedef __attribute__((ext_vector_type(16))) _Float16 v16h;
typedef __attribute__((ext_vector_type(8)))  float    v8f;
typedef __attribute__((ext_vector_type(4)))  int      v4i;
typedef __attribute__((ext_vector_type(4)))  unsigned v4u;

#define NB    32
#define CIN   512
#define COUT  512
#define RES   56
#define NHD   8
#define HD    64
#define NTOK  49
#define NPIX  (RES*RES)        // 3136
#define ROWS  (NB*NTOK)        // 1568

// ---------------------------------------------------------------------------
// WMMA 16x16x32 f16 fragment loader for a row-major matrix [row][k].
// Works for both A (row = M) and B (row = N, since B[k][n] = W[n][k]).
// 16-bit A/B layout: lane l -> row = base + (l&15); lanes 16-31 take K+8;
// VGPR j in 0..3 -> K pairs {2j,2j+1}; VGPR 4..7 -> same +16.
// Per lane that is two contiguous 16B runs -> two b128 loads (ds or global).
// All call sites guarantee 16B alignment (row strides 1024/1040/1056 B,
// k0 multiples of 64 B, lane-half offset 16 B).
// ---------------------------------------------------------------------------
__device__ __forceinline__ v16h frag_ld(const _Float16* m, int ld, int rowbase, int k0) {
  int l = threadIdx.x & 31;
  const _Float16* pr = m + (size_t)(rowbase + (l & 15)) * ld + (k0 + ((l >> 4) << 3));
  union { v16h v; v4u q[2]; } U;
  U.q[0] = *(const v4u*)pr;          // K pairs 0,2,4,6
  U.q[1] = *(const v4u*)(pr + 16);   // K pairs 16,18,20,22
  return U.v;
}

#define WMMA(a, b, c) __builtin_amdgcn_wmma_f32_16x16x32_f16(false, a, false, b, (short)0, c, false, false)

// ---------------------------------------------------------------------------
// Async global->LDS 16-byte copy (CDNA5 GLOBAL_LOAD_ASYNC_TO_LDS_B128),
// with synchronous fallback if the builtin is unavailable.
// Builtin prototype (from hipcc diagnostic): (v4i AS1*, v4i AS3*, imm, imm).
// ---------------------------------------------------------------------------
#if __has_builtin(__builtin_amdgcn_global_load_async_to_lds_b128)
#define HAVE_ASYNC_LDS 1
#else
#define HAVE_ASYNC_LDS 0
#endif

__device__ __forceinline__ void async_cp16(const _Float16* g, _Float16* l) {
#if HAVE_ASYNC_LDS
  __builtin_amdgcn_global_load_async_to_lds_b128(
      (__attribute__((address_space(1))) v4i*)g,
      (__attribute__((address_space(3))) v4i*)l, 0, 0);
#else
  *(float4*)l = *(const float4*)g;
#endif
}

__device__ __forceinline__ void async_wait_all() {
#if HAVE_ASYNC_LDS
  asm volatile("s_wait_asynccnt 0x0" ::: "memory");
#endif
}

// ---------------------------------------------------------------------------
// f32 -> f16 weight conversion
// ---------------------------------------------------------------------------
__global__ void k_cvt(const float* __restrict__ s, _Float16* __restrict__ d, int n) {
  int i = blockIdx.x * 256 + threadIdx.x;
  if (i < n) d[i] = (_Float16)s[i];
}

// ---------------------------------------------------------------------------
// Depthwise 3x3 stride-8 conv + BN -> xs_t[b*49+p][c] (f16, row-major in c)
// ---------------------------------------------------------------------------
__global__ __launch_bounds__(256) void k_dw(const float* __restrict__ x,
                                            const float* __restrict__ dw,
                                            const float* __restrict__ s0,
                                            const float* __restrict__ b0,
                                            _Float16* __restrict__ xs_t) {
  int bp = blockIdx.x;                 // 0..1567
  int b = bp / NTOK, p = bp % NTOK;
  int py = p / 7, px = p % 7;
  for (int c = threadIdx.x; c < CIN; c += 256) {
    const float* xc = x + ((size_t)b * CIN + c) * NPIX;
    const float* w  = dw + c * 9;
    float acc = 0.f;
#pragma unroll
    for (int ky = 0; ky < 3; ++ky) {
      int iy = py * 8 + ky - 1;
      if (iy < 0 || iy >= RES) continue;
#pragma unroll
      for (int kx = 0; kx < 3; ++kx) {
        int ix = px * 8 + kx - 1;
        if (ix < 0 || ix >= RES) continue;
        acc += w[ky * 3 + kx] * xc[iy * RES + ix];
      }
    }
    xs_t[(size_t)bp * CIN + c] = (_Float16)(acc * s0[c] + b0[c]);
  }
}

// ---------------------------------------------------------------------------
// q/k/v 1x1 convs as WMMA GEMM: Y[bp][o] = BN( xs_t[bp][:] . W[o][:] )
// grid (25 m-blocks of 64 rows, 3 matrices), 256 threads = 8 waves.
// A tile async-staged into LDS; B fragment loaded once and reused by 4 WMMAs.
// ---------------------------------------------------------------------------
#define LDQ 528   // 64-row A tile stride in halfs (1056B rows -> conflict-free)
__global__ __launch_bounds__(256) void k_qkv(const _Float16* __restrict__ xs_t,
                                             const _Float16* __restrict__ wq_h,
                                             const _Float16* __restrict__ wk_h,
                                             const _Float16* __restrict__ wv_h,
                                             const float* __restrict__ sq, const float* __restrict__ bq,
                                             const float* __restrict__ sk, const float* __restrict__ bk,
                                             const float* __restrict__ sv, const float* __restrict__ bv,
                                             float* __restrict__ q_y,
                                             float* __restrict__ k_y,
                                             float* __restrict__ v_y) {
  extern __shared__ _Float16 As[];               // [64][LDQ]
  const _Float16* W; const float* s; const float* bb; float* y;
  if (blockIdx.y == 0)      { W = wq_h; s = sq; bb = bq; y = q_y; }
  else if (blockIdx.y == 1) { W = wk_h; s = sk; bb = bk; y = k_y; }
  else                      { W = wv_h; s = sv; bb = bv; y = v_y; }
  int wav = threadIdx.x >> 5;
  int l   = threadIdx.x & 31;
  int mbase = blockIdx.x * 64;

  // stage A tile: thread t -> row t>>2, quarter (t&3) of 1024B row, 16x16B chunks
  {
    int row = threadIdx.x >> 2;
    int hoff = (threadIdx.x & 3) * 128;          // halfs
    int grow = mbase + row;
    if (grow < ROWS) {
      const _Float16* g = xs_t + (size_t)grow * CIN + hoff;
      _Float16* lp = As + row * LDQ + hoff;
#pragma unroll
      for (int c2 = 0; c2 < 16; ++c2) async_cp16(g + c2 * 8, lp + c2 * 8);
    }
  }
  async_wait_all();
  __syncthreads();

  for (int nt = wav; nt < 32; nt += 8) {
    v8f a0 = {}, a1 = {}, a2 = {}, a3 = {};
    for (int kk = 0; kk < 16; ++kk) {
      v16h bf = frag_ld(W, CIN, nt * 16, kk * 32);     // one B load ...
      v16h f0 = frag_ld(As, LDQ,  0, kk * 32);
      v16h f1 = frag_ld(As, LDQ, 16, kk * 32);
      v16h f2 = frag_ld(As, LDQ, 32, kk * 32);
      v16h f3 = frag_ld(As, LDQ, 48, kk * 32);
      a0 = WMMA(f0, bf, a0);                           // ... feeds 4 WMMAs
      a1 = WMMA(f1, bf, a1);
      a2 = WMMA(f2, bf, a2);
      a3 = WMMA(f3, bf, a3);
    }
    int o = nt * 16 + (l & 15);
    float sc = s[o], bc = bb[o];
#pragma unroll
    for (int r = 0; r < 8; ++r) {
      int rr = r + ((l >> 4) << 3);
      int r0 = mbase + rr,      r1 = mbase + 16 + rr;
      int r2 = mbase + 32 + rr, r3 = mbase + 48 + rr;
      if (r0 < ROWS) y[(size_t)r0 * COUT + o] = sc * a0[r] + bc;
      if (r1 < ROWS) y[(size_t)r1 * COUT + o] = sc * a1[r] + bc;
      if (r2 < ROWS) y[(size_t)r2 * COUT + o] = sc * a2[r] + bc;
      if (r3 < ROWS) y[(size_t)r3 * COUT + o] = sc * a3[r] + bc;
    }
  }
}

// ---------------------------------------------------------------------------
// v_local: grouped 3x3 conv (groups=8) + bias + BN.  block = (b, h).
// ---------------------------------------------------------------------------
__global__ __launch_bounds__(256) void k_vloc(const float* __restrict__ v_y,
                                              const float* __restrict__ lw,
                                              const float* __restrict__ lb,
                                              const float* __restrict__ sl,
                                              const float* __restrict__ bl,
                                              float* __restrict__ vloc) {
  int b = blockIdx.x, h = blockIdx.y;
  __shared__ float vg[NTOK * HD];
  for (int i = threadIdx.x; i < NTOK * HD; i += 256) {
    int p = i >> 6, d = i & 63;
    vg[i] = v_y[((size_t)b * NTOK + p) * COUT + h * HD + d];
  }
  __syncthreads();
  for (int i = threadIdx.x; i < NTOK * HD; i += 256) {
    int p = i >> 6, od = i & 63;
    int py = p / 7, px = p % 7;
    int o = h * HD + od;
    const float* w = lw + (size_t)o * HD * 9;      // loc_w [512][64][3][3]
    float acc = 0.f;
#pragma unroll
    for (int ky = 0; ky < 3; ++ky) {
      int iy = py + ky - 1;
      if (iy < 0 || iy >= 7) continue;
#pragma unroll
      for (int kx = 0; kx < 3; ++kx) {
        int ix = px + kx - 1;
        if (ix < 0 || ix >= 7) continue;
        const float* vp = vg + (iy * 7 + ix) * HD;
        const float* wp = w + (ky * 3 + kx);
        for (int di = 0; di < HD; ++di) acc += wp[di * 9] * vp[di];
      }
    }
    acc += lb[o];
    vloc[((size_t)b * NTOK + p) * COUT + o] = acc * sl[o] + bl[o];
  }
}

// ---------------------------------------------------------------------------
// Attention: cosine logits + pos-bias, talking-heads 1, softmax,
// talking-heads 2, @v, + v_local  -> ff16[bp][c].  One block per batch,
// wave h owns head h; cross-head mixing via LDS (2 x [8][49][49] f32).
// ---------------------------------------------------------------------------
__global__ __launch_bounds__(256) void k_attn(const float* __restrict__ q_y,
                                              const float* __restrict__ k_y,
                                              const float* __restrict__ v_y,
                                              const float* __restrict__ vloc,
                                              const float* __restrict__ th1,
                                              const float* __restrict__ th2,
                                              const float* __restrict__ lsc,
                                              const float* __restrict__ ab,
                                              const int*   __restrict__ bias_idx,
                                              _Float16* __restrict__ ff) {
  extern __shared__ float smem[];
  float* L    = smem;                       // [8][49][49]
  float* S    = L + NHD * NTOK * NTOK;      // [8][49][49]
  float* invk = S + NHD * NTOK * NTOK;      // [8][64]
  int b = blockIdx.x;
  int h = threadIdx.x >> 5;
  int l = threadIdx.x & 31;

  // 1/||k[d,:]|| over tokens (axis -1 of [B,H,d,N])
  for (int d = l; d < HD; d += 32) {
    float s = 0.f;
    for (int m = 0; m < NTOK; ++m) {
      float kv = k_y[((size_t)b * NTOK + m) * COUT + h * HD + d];
      s += kv * kv;
    }
    invk[h * HD + d] = 1.f / fmaxf(sqrtf(s), 1e-12f);
  }
  __syncthreads();

  float ls = __expf(fminf(lsc[h], __logf(100.f)));
  for (int n = l; n < NTOK; n += 32) {
    const float* qr = q_y + ((size_t)b * NTOK + n) * COUT + h * HD;
    float qs = 0.f;
    for (int d = 0; d < HD; ++d) { float qv = qr[d]; qs += qv * qv; }
    float qinv = 1.f / fmaxf(sqrtf(qs), 1e-12f);
    const float* iv = invk + h * HD;
    for (int m = 0; m < NTOK; ++m) {
      const float* kr = k_y + ((size_t)b * NTOK + m) * COUT + h * HD;
      float acc = 0.f;
      for (int d = 0; d < HD; ++d) acc += qr[d] * kr[d] * iv[d];
      L[(h * NTOK + n) * NTOK + m] =
          acc * qinv * ls + ab[h * NTOK + bias_idx[n * NTOK + m]];
    }
  }
  __syncthreads();

  // talking-head 1 + softmax
  for (int n = l; n < NTOK; n += 32) {
    float row[NTOK];
    for (int m = 0; m < NTOK; ++m) {
      float acc = 0.f;
#pragma unroll
      for (int j = 0; j < NHD; ++j) acc += th1[h * NHD + j] * L[(j * NTOK + n) * NTOK + m];
      row[m] = acc;
    }
    float mx = row[0];
    for (int m = 1; m < NTOK; ++m) mx = fmaxf(mx, row[m]);
    float sum = 0.f;
    for (int m = 0; m < NTOK; ++m) { row[m] = __expf(row[m] - mx); sum += row[m]; }
    float inv = 1.f / sum;
    for (int m = 0; m < NTOK; ++m) S[(h * NTOK + n) * NTOK + m] = row[m] * inv;
  }
  __syncthreads();

  // talking-head 2, @v, + v_local
  for (int n = l; n < NTOK; n += 32) {
    float a2[NTOK];
    for (int m = 0; m < NTOK; ++m) {
      float acc = 0.f;
#pragma unroll
      for (int j = 0; j < NHD; ++j) acc += th2[h * NHD + j] * S[(j * NTOK + n) * NTOK + m];
      a2[m] = acc;
    }
    for (int d = 0; d < HD; ++d) {
      float acc = 0.f;
      for (int m = 0; m < NTOK; ++m)
        acc += a2[m] * v_y[((size_t)b * NTOK + m) * COUT + h * HD + d];
      size_t idx = ((size_t)b * NTOK + n) * COUT + h * HD + d;
      ff[idx] = (_Float16)(acc + vloc[idx]);
    }
  }
}

// ---------------------------------------------------------------------------
// Fused: async-stage ff[b] slab (49x512 f16) to LDS, bilinear 7x7 -> 56x56
// upsample + Hardswish -> LDS f16 A-tile, then 64x512x512 WMMA GEMM with
// wo + BN -> out [32,512,56,56] f32.  grid = (32, 49 pixel-tiles of 64).
// ---------------------------------------------------------------------------
#define LDA 520
#define SLAB (NTOK * COUT)                 // 25088 halfs
__global__ __launch_bounds__(256) void k_out(const _Float16* __restrict__ ff,
                                             const _Float16* __restrict__ wo_h,
                                             const float* __restrict__ so,
                                             const float* __restrict__ bo,
                                             float* __restrict__ out) {
  extern __shared__ _Float16 sh[];
  _Float16* Sl = sh;                       // [49][512] pre-upsample slab
  _Float16* A  = sh + SLAB;                // [64][LDA]
  int b = blockIdx.x, tile = blockIdx.y;
  int l = threadIdx.x & 31;
  int wav = threadIdx.x >> 5;

  // async copy the whole per-batch slab into LDS (3136 x 16B chunks)
  {
    const _Float16* g = ff + (size_t)b * SLAB;
    for (int c2 = threadIdx.x; c2 < SLAB / 8; c2 += 256)
      async_cp16(g + c2 * 8, Sl + c2 * 8);
  }
  async_wait_all();
  __syncthreads();

  // Build A: thread t -> pixel i = t>>2, channels [(t&3)*128, +128)
  {
    int i = threadIdx.x >> 2;
    int c0 = (threadIdx.x & 3) * 128;
    int gp = tile * 64 + i;
    int y = gp / RES, x = gp % RES;
    float sy = (y + 0.5f) * 0.125f - 0.5f;
    float sx = (x + 0.5f) * 0.125f - 0.5f;
    int y0 = (int)floorf(sy); float wy = sy - y0;
    int x0 = (int)floorf(sx); float wx = sx - x0;
    int y1 = y0 + 1 > 6 ? 6 : y0 + 1;  y0 = y0 < 0 ? 0 : y0;
    int x1 = x0 + 1 > 6 ? 6 : x0 + 1;  x0 = x0 < 0 ? 0 : x0;
    const _Float16* r00 = Sl + (y0 * 7 + x0) * COUT;
    const _Float16* r01 = Sl + (y0 * 7 + x1) * COUT;
    const _Float16* r10 = Sl + (y1 * 7 + x0) * COUT;
    const _Float16* r11 = Sl + (y1 * 7 + x1) * COUT;
    float w00 = (1.f - wy) * (1.f - wx), w01 = (1.f - wy) * wx;
    float w10 = wy * (1.f - wx),         w11 = wy * wx;
    for (int c = c0; c < c0 + 128; ++c) {
      float u = w00 * (float)r00[c] + w01 * (float)r01[c]
              + w10 * (float)r10[c] + w11 * (float)r11[c];
      float t6 = u + 3.f; t6 = t6 < 0.f ? 0.f : (t6 > 6.f ? 6.f : t6);
      A[i * LDA + c] = (_Float16)(u * t6 * (1.f / 6.f));
    }
  }
  __syncthreads();

  for (int nt = wav; nt < 32; nt += 8) {
    v8f a0 = {}, a1 = {}, a2 = {}, a3 = {};
    for (int kk = 0; kk < 16; ++kk) {
      v16h bf = frag_ld(wo_h, COUT, nt * 16, kk * 32);  // one B load ...
      v16h f0 = frag_ld(A, LDA,  0, kk * 32);
      v16h f1 = frag_ld(A, LDA, 16, kk * 32);
      v16h f2 = frag_ld(A, LDA, 32, kk * 32);
      v16h f3 = frag_ld(A, LDA, 48, kk * 32);
      a0 = WMMA(f0, bf, a0);                            // ... feeds 4 WMMAs
      a1 = WMMA(f1, bf, a1);
      a2 = WMMA(f2, bf, a2);
      a3 = WMMA(f3, bf, a3);
    }
    int o = nt * 16 + (l & 15);
    float sc = so[o], bc = bo[o];
    float* ob = out + ((size_t)b * COUT + o) * NPIX + tile * 64;
#pragma unroll
    for (int r = 0; r < 8; ++r) {
      int rr = r + ((l >> 4) << 3);
      ob[rr]      = sc * a0[r] + bc;
      ob[rr + 16] = sc * a1[r] + bc;
      ob[rr + 32] = sc * a2[r] + bc;
      ob[rr + 48] = sc * a3[r] + bc;
    }
  }
}

// ---------------------------------------------------------------------------
extern "C" void kernel_launch(void* const* d_in, const int* in_sizes, int n_in,
                              void* d_out, int out_size, void* d_ws, size_t ws_size,
                              hipStream_t stream) {
  (void)in_sizes; (void)n_in; (void)out_size; (void)ws_size;
  const float* x      = (const float*)d_in[0];
  const float* dw_w   = (const float*)d_in[1];
  const float* bn0_s  = (const float*)d_in[2];
  const float* bn0_b  = (const float*)d_in[3];
  const float* wq     = (const float*)d_in[4];
  const float* bnq_s  = (const float*)d_in[5];
  const float* bnq_b  = (const float*)d_in[6];
  const float* wk     = (const float*)d_in[7];
  const float* bnk_s  = (const float*)d_in[8];
  const float* bnk_b  = (const float*)d_in[9];
  const float* wv     = (const float*)d_in[10];
  const float* bnv_s  = (const float*)d_in[11];
  const float* bnv_b  = (const float*)d_in[12];
  const float* loc_w  = (const float*)d_in[13];
  const float* loc_b  = (const float*)d_in[14];
  const float* bnl_s  = (const float*)d_in[15];
  const float* bnl_b  = (const float*)d_in[16];
  const float* th1    = (const float*)d_in[17];
  const float* th2    = (const float*)d_in[18];
  const float* lsc    = (const float*)d_in[19];
  const float* ab     = (const float*)d_in[20];
  const float* wo     = (const float*)d_in[21];
  const float* bno_s  = (const float*)d_in[22];
  const float* bno_b  = (const float*)d_in[23];
  const int*   bidx   = (const int*)d_in[24];
  float* out = (float*)d_out;

  // workspace carve-up
  char* w = (char*)d_ws;
  auto take = [&](size_t bytes) { char* p = w; w += (bytes + 255) & ~(size_t)255; return p; };
  _Float16* xs_t = (_Float16*)take((size_t)ROWS * CIN * 2);
  _Float16* wq_h = (_Float16*)take((size_t)COUT * CIN * 2);
  _Float16* wk_h = (_Float16*)take((size_t)COUT * CIN * 2);
  _Float16* wv_h = (_Float16*)take((size_t)COUT * CIN * 2);
  _Float16* wo_h = (_Float16*)take((size_t)COUT * COUT * 2);
  float* q_y  = (float*)take((size_t)ROWS * COUT * 4);
  float* k_y  = (float*)take((size_t)ROWS * COUT * 4);
  float* v_y  = (float*)take((size_t)ROWS * COUT * 4);
  float* vloc = (float*)take((size_t)ROWS * COUT * 4);
  _Float16* ff = (_Float16*)take((size_t)ROWS * COUT * 2);

  const int NW = COUT * CIN;           // 262144
  k_cvt<<<(NW + 255) / 256, 256, 0, stream>>>(wq, wq_h, NW);
  k_cvt<<<(NW + 255) / 256, 256, 0, stream>>>(wk, wk_h, NW);
  k_cvt<<<(NW + 255) / 256, 256, 0, stream>>>(wv, wv_h, NW);
  k_cvt<<<(NW + 255) / 256, 256, 0, stream>>>(wo, wo_h, NW);

  k_dw<<<ROWS, 256, 0, stream>>>(x, dw_w, bn0_s, bn0_b, xs_t);

  size_t qkv_smem = (size_t)64 * LDQ * sizeof(_Float16);
  k_qkv<<<dim3(25, 3), 256, qkv_smem, stream>>>(xs_t, wq_h, wk_h, wv_h,
                                                bnq_s, bnq_b, bnk_s, bnk_b, bnv_s, bnv_b,
                                                q_y, k_y, v_y);

  k_vloc<<<dim3(NB, NHD), 256, 0, stream>>>(v_y, loc_w, loc_b, bnl_s, bnl_b, vloc);

  size_t attn_smem = (size_t)(2 * NHD * NTOK * NTOK + NHD * HD) * sizeof(float);
  k_attn<<<NB, 256, attn_smem, stream>>>(q_y, k_y, v_y, vloc, th1, th2, lsc, ab, bidx, ff);

  size_t out_smem = (size_t)(SLAB + 64 * LDA) * sizeof(_Float16);
  k_out<<<dim3(NB, 49), 256, out_smem, stream>>>(ff, wo_h, bno_s, bno_b, out);
}
